// DeepCAD_49331994361927
// MI455X (gfx1250) — matrix operand order repaired and checked by reading, not compiled, not run on hardware
//
#include <hip/hip_runtime.h>
#include <hip/hip_bf16.h>

typedef _Float16 half_t;
typedef __attribute__((ext_vector_type(16))) _Float16 v16h;
typedef __attribute__((ext_vector_type(8)))  float    v8f;

#define WAVES_PER_BLOCK 8
#define BLOCK_THREADS   256

__device__ __forceinline__ v8f wmma_f16(v16h a, v16h b, v8f c) {
    // D = A(16x32 f16) * B(32x16 f16) + C(16x16 f32)
    return __builtin_amdgcn_wmma_f32_16x16x32_f16(
        /*neg_a=*/false, a, /*neg_b=*/false, b,
        /*c_mod=*/(short)0, c, /*reuse_a=*/false, /*reuse_b=*/false);
}

// A-fragment gather from an LDS f16 tile with row stride ldA (rows = M, cols = K-chunk of 32).
// 16-bit A 16x32 layout (contiguous K, matching B): lanes 0-15 hold K 0..15 (half i = K i);
// lanes 16-31 hold K 16..31. Row M = lane & 15.
__device__ __forceinline__ v16h load_fragA_lds(const half_t* As, int ldA) {
    int lane = threadIdx.x & 31;
    int m  = lane & 15;
    int kb = (lane >> 4) << 4;          // 0 or 16
    v16h a;
#pragma unroll
    for (int i = 0; i < 8; ++i) a[i]     = As[m * ldA + kb + i];
#pragma unroll
    for (int i = 0; i < 8; ++i) a[8 + i] = As[m * ldA + kb + 8 + i];
    return a;
}

// B-fragment gather from an LDS f16 tile (K=32 rows, ldB cols), columns ncol0..ncol0+15.
// 16-bit B 32x16, contiguous K (matching A): lanes 0-15 hold K=0..15 (half i = K i), lanes 16-31 hold K=16..31. N = lane & 15.
__device__ __forceinline__ v16h load_fragB_lds(const half_t* Bs, int ldB, int ncol0) {
    int lane = threadIdx.x & 31;
    int n  = ncol0 + (lane & 15);
    int kb = (lane >> 4) << 4;          // 0 or 16
    v16h b;
#pragma unroll
    for (int i = 0; i < 16; ++i) b[i] = Bs[(kb + i) * ldB + n];
    return b;
}

// Packed B fragments in global memory: frag f occupies 512 halfs, lane-major (lane*16 contiguous).
__device__ __forceinline__ v16h load_fragB_packed(const half_t* P, int f) {
    return ((const v16h*)(P + ((size_t)f << 9)))[threadIdx.x & 31];
}

__device__ __forceinline__ float sigf(float x) { return 1.0f / (1.0f + __expf(-x)); }

// ---------------------------------------------------------------------------
// Repack a GEMM weight into f16 B-fragment order.
// transpose==0: W is [K, N] row-major, B[k][n] = W[k][n]   (x @ W)
// transpose==1: W is [N, K] row-major, B[k][n] = W[n][k]   (x @ W^T)
// One thread per (fragment, lane); writes 16 contiguous halfs.
__global__ void repack_frags(const float* __restrict__ W, half_t* __restrict__ P,
                             int K, int N, int transpose) {
    int KT = K >> 5, NT = N >> 4;
    int idx = blockIdx.x * blockDim.x + threadIdx.x;
    int total = KT * NT * 32;
    if (idx >= total) return;
    int lane = idx & 31;
    int f    = idx >> 5;
    int nt   = f % NT, kt = f / NT;
    int n  = (nt << 4) + (lane & 15);
    int kb = (kt << 5) + ((lane >> 4) << 4);
    half_t* out = P + ((size_t)f << 9) + lane * 16;
#pragma unroll
    for (int i = 0; i < 16; ++i) {
        int k = kb + i;
        float v = transpose ? W[(size_t)n * K + k] : W[(size_t)k * N + n];
        out[i] = (half_t)v;
    }
}

// ---------------------------------------------------------------------------
// Generic GEMM: C[M,N] = act(A[M,K] @ Bpacked + bias). K in {64,128}, N multiple of 16.
// grid.x = M/16, 256 threads = 8 waves; wave w handles N-tiles w, w+8, ...
__global__ void gemm_bias_act(const float* __restrict__ A, const half_t* __restrict__ Bp,
                              const float* __restrict__ bias, float* __restrict__ C,
                              int M, int K, int N, int act) {
    __shared__ half_t As[16 * 128];
    int tid = threadIdx.x;
    int m0  = blockIdx.x << 4;
    for (int e = tid; e < 16 * K; e += BLOCK_THREADS) {
        int r = e / K, c = e - r * K;
        As[r * K + c] = (half_t)A[(size_t)(m0 + r) * K + c];
    }
    __syncthreads();
    int wave = tid >> 5, lane = tid & 31;
    int KT = K >> 5, NT = N >> 4;
    for (int nt = wave; nt < NT; nt += WAVES_PER_BLOCK) {
        v8f acc = {};
        for (int kt = 0; kt < KT; ++kt) {
            v16h a = load_fragA_lds(As + kt * 32, K);
            v16h b = load_fragB_packed(Bp, kt * NT + nt);
            acc = wmma_f16(a, b, acc);
        }
        int n  = (nt << 4) + (lane & 15);
        int mo = (lane >> 4) << 3;
        float bv = bias ? bias[n] : 0.0f;
#pragma unroll
        for (int r = 0; r < 8; ++r) {
            float v = acc[r] + bv;
            if (act) v = fmaxf(v, 0.0f);
            C[(size_t)(m0 + mo + r) * N + n] = v;
        }
    }
}

// ---------------------------------------------------------------------------
// Attention score projections: sj = h @ aW[:H], si = h @ aW[H:2H]   (per flattened row)
__global__ void gnn_scores(const float* __restrict__ h, const float* __restrict__ aW,
                           float* __restrict__ si, float* __restrict__ sj, int M, int H) {
    int row = blockIdx.x * blockDim.x + threadIdx.x;
    if (row >= M) return;
    const float* hr = h + (size_t)row * H;
    float s0 = 0.0f, s1 = 0.0f;
    for (int k = 0; k < H; ++k) {
        float v = hr[k];
        s0 += v * aW[k];        // h_j term
        s1 += v * aW[H + k];    // h_i term
    }
    sj[row] = s0;
    si[row] = s1;
}

// ---------------------------------------------------------------------------
// Fused attention + aggregation:
// out[b,i,:] = sum_j sigmoid(si[b,i] + sj[b,j] + ab) * adj[b,i,j] * h[b,j,:]
// grid = B * (N/32) blocks, 256 threads (8 waves).
// Block computes a 32(i) x 128(n) output tile; j staged in chunks of 64.
// Wave w owns columns w*16..w*16+15 and both 16-row i-subtiles (B fragment reused).
__global__ void gnn_aggregate(const float* __restrict__ hfeat, const float* __restrict__ adj,
                              const float* __restrict__ si, const float* __restrict__ sj,
                              const float* __restrict__ ab, float* __restrict__ out, int act) {
    __shared__ half_t As[32 * 64];     // f16 attention tile  [i][j]   4 KB
    __shared__ half_t Bs[64 * 128];    // f16 h tile          [j][n]  16 KB
    __shared__ float  s_si[32];
    const int Nn = 1024, H = 128;
    int blk = blockIdx.x;
    int b   = blk >> 5;                // Nn/32 == 32 blocks per batch
    int i0  = (blk & 31) << 5;
    int tid = threadIdx.x, wave = tid >> 5, lane = tid & 31;
    float abv = ab[0];
    if (tid < 32) s_si[tid] = si[b * Nn + i0 + tid];
    __syncthreads();
    const float* adjb = adj   + (size_t)b * Nn * Nn;
    const float* hb   = hfeat + (size_t)b * Nn * H;
    const float* sjb  = sj + b * Nn;
    v8f acc0 = {}, acc1 = {};
    for (int j0 = 0; j0 < Nn; j0 += 64) {
        // prefetch next adjacency stripe (global_prefetch_b8)
        if (j0 + 64 < Nn)
            __builtin_prefetch(&adjb[(size_t)(i0 + (tid & 31)) * Nn + j0 + 64], 0, 0);
        // stage attention tile (32 x 64)
        for (int e = tid; e < 2048; e += BLOCK_THREADS) {
            int r = e >> 6, c = e & 63;
            int j = j0 + c;
            float g = sigf(s_si[r] + sjb[j] + abv);
            As[e] = (half_t)(g * adjb[(size_t)(i0 + r) * Nn + j]);
        }
        // stage h tile (64 x 128)
        for (int e = tid; e < 8192; e += BLOCK_THREADS) {
            int k = e >> 7, n = e & 127;
            Bs[e] = (half_t)hb[(size_t)(j0 + k) * H + n];
        }
        __syncthreads();
#pragma unroll
        for (int kk = 0; kk < 2; ++kk) {          // two K=32 steps in the 64-chunk
            v16h bb = load_fragB_lds(Bs + (kk << 5) * H, H, wave << 4);
            v16h a0 = load_fragA_lds(As + (kk << 5), 64);
            acc0 = wmma_f16(a0, bb, acc0);
            v16h a1 = load_fragA_lds(As + 16 * 64 + (kk << 5), 64);
            acc1 = wmma_f16(a1, bb, acc1);
        }
        __syncthreads();
    }
    int n  = (wave << 4) + (lane & 15);
    int mo = (lane >> 4) << 3;
    float* ob = out + (size_t)b * Nn * H;
#pragma unroll
    for (int r = 0; r < 8; ++r) {
        float v0 = acc0[r], v1 = acc1[r];
        if (act) { v0 = fmaxf(v0, 0.0f); v1 = fmaxf(v1, 0.0f); }
        ob[(size_t)(i0 + mo + r) * H + n]      = v0;
        ob[(size_t)(i0 + 16 + mo + r) * H + n] = v1;
    }
}

// ---------------------------------------------------------------------------
// One LSTM layer (torch gate order i,f,g,o), zero init state. x,y: [8,1024,128].
// Single persistent block, 256 threads (8 waves); batch 8 padded to M=16.
__global__ void lstm_layer(const float* __restrict__ x,
                           const half_t* __restrict__ WihP, const half_t* __restrict__ WhhP,
                           const float* __restrict__ bih, const float* __restrict__ bhh,
                           float* __restrict__ y) {
    __shared__ half_t Xs[16 * 128];
    __shared__ half_t Hs[16 * 128];
    __shared__ float  Gs[16 * 512];
    __shared__ float  hS[8 * 128];
    __shared__ float  cS[8 * 128];
    const int T = 1024, H = 128;
    int tid = threadIdx.x, wave = tid >> 5, lane = tid & 31;
    for (int e = tid; e < 8 * H; e += BLOCK_THREADS) { hS[e] = 0.0f; cS[e] = 0.0f; }
    for (int e = tid; e < 16 * H; e += BLOCK_THREADS) { Xs[e] = (half_t)0.0f; Hs[e] = (half_t)0.0f; }
    __syncthreads();
    const int KT = 4, NT = 32;  // K=128, Ngates=512
    for (int t = 0; t < T; ++t) {
        for (int e = tid; e < 8 * H; e += BLOCK_THREADS) {
            int r = e >> 7, c = e & 127;
            Xs[r * H + c] = (half_t)x[((size_t)r * T + t) * H + c];
            Hs[r * H + c] = (half_t)hS[e];
        }
        __syncthreads();
#pragma unroll
        for (int q = 0; q < 4; ++q) {
            int nt = (wave << 2) + q;
            v8f acc = {};
#pragma unroll
            for (int kt = 0; kt < KT; ++kt) {
                v16h a = load_fragA_lds(Xs + kt * 32, H);
                v16h b = load_fragB_packed(WihP, kt * NT + nt);
                acc = wmma_f16(a, b, acc);
            }
#pragma unroll
            for (int kt = 0; kt < KT; ++kt) {
                v16h a = load_fragA_lds(Hs + kt * 32, H);
                v16h b = load_fragB_packed(WhhP, kt * NT + nt);
                acc = wmma_f16(a, b, acc);
            }
            int n  = (nt << 4) + (lane & 15);
            int mo = (lane >> 4) << 3;
#pragma unroll
            for (int r = 0; r < 8; ++r) Gs[(mo + r) * 512 + n] = acc[r];
        }
        __syncthreads();
        for (int e = tid; e < 8 * H; e += BLOCK_THREADS) {
            int r = e >> 7, c = e & 127;
            float ig = Gs[r * 512 +   0 + c] + bih[  0 + c] + bhh[  0 + c];
            float fg = Gs[r * 512 + 128 + c] + bih[128 + c] + bhh[128 + c];
            float gg = Gs[r * 512 + 256 + c] + bih[256 + c] + bhh[256 + c];
            float og = Gs[r * 512 + 384 + c] + bih[384 + c] + bhh[384 + c];
            float cc = sigf(fg) * cS[e] + sigf(ig) * tanhf(gg);
            float hh = sigf(og) * tanhf(cc);
            cS[e] = cc; hS[e] = hh;
            y[((size_t)r * T + t) * H + c] = hh;
        }
        __syncthreads();
    }
}

// ---------------------------------------------------------------------------
extern "C" void kernel_launch(void* const* d_in, const int* in_sizes, int n_in,
                              void* d_out, int out_size, void* d_ws, size_t ws_size,
                              hipStream_t stream) {
    const int Bb = 8, Nn = 1024, H = 128, M = Bb * Nn;  // 8192 rows
    const float* node = (const float*)d_in[0];
    const float* adj  = (const float*)d_in[1];
    // d_in[2] = mask (unused by the reference)
    const float* g_W[3]  = { (const float*)d_in[3],  (const float*)d_in[7],  (const float*)d_in[11] };
    const float* g_b[3]  = { (const float*)d_in[4],  (const float*)d_in[8],  (const float*)d_in[12] };
    const float* g_aW[3] = { (const float*)d_in[5],  (const float*)d_in[9],  (const float*)d_in[13] };
    const float* g_ab[3] = { (const float*)d_in[6],  (const float*)d_in[10], (const float*)d_in[14] };
    const float* l_Wih[2] = { (const float*)d_in[15], (const float*)d_in[19] };
    const float* l_Whh[2] = { (const float*)d_in[16], (const float*)d_in[20] };
    const float* l_bih[2] = { (const float*)d_in[17], (const float*)d_in[21] };
    const float* l_bhh[2] = { (const float*)d_in[18], (const float*)d_in[22] };
    const float* hW1[4] = { (const float*)d_in[23], (const float*)d_in[27], (const float*)d_in[31], (const float*)d_in[35] };
    const float* hb1[4] = { (const float*)d_in[24], (const float*)d_in[28], (const float*)d_in[32], (const float*)d_in[36] };
    const float* hW2[4] = { (const float*)d_in[25], (const float*)d_in[29], (const float*)d_in[33], (const float*)d_in[37] };
    const float* hb2[4] = { (const float*)d_in[26], (const float*)d_in[30], (const float*)d_in[34], (const float*)d_in[38] };
    const int hd2[4] = { 64, 256, 128, 64 };   // op, pa, sk, no output dims

    // ---- workspace carve ----
    char* ws = (char*)d_ws;
    const size_t BUF = (size_t)M * H * sizeof(float);       // 4 MB
    float* bufA = (float*)(ws);
    float* bufB = (float*)(ws + BUF);
    float* bufC = (float*)(ws + 2 * BUF);
    float* si   = (float*)(ws + 3 * BUF);
    float* sj   = si + M;
    half_t* pk  = (half_t*)(ws + 3 * BUF + 64 * 1024);
    auto alloc_pk = [&](int K, int Nout) { half_t* p = pk; pk += (size_t)K * Nout; return p; };

    auto repack = [&](const float* W, half_t* P, int K, int Nout, int tr) {
        int total = (K >> 5) * (Nout >> 4) * 32;
        repack_frags<<<(total + 255) / 256, 256, 0, stream>>>(W, P, K, Nout, tr);
    };

    half_t* pgW[3];
    pgW[0] = alloc_pk(64, H);  repack(g_W[0], pgW[0], 64, H, 0);
    pgW[1] = alloc_pk(H, H);   repack(g_W[1], pgW[1], H, H, 0);
    pgW[2] = alloc_pk(H, H);   repack(g_W[2], pgW[2], H, H, 0);
    half_t *pLih[2], *pLhh[2];
    for (int l = 0; l < 2; ++l) {
        pLih[l] = alloc_pk(H, 4 * H); repack(l_Wih[l], pLih[l], H, 4 * H, 1);  // x @ Wih^T
        pLhh[l] = alloc_pk(H, 4 * H); repack(l_Whh[l], pLhh[l], H, 4 * H, 1);
    }
    half_t *pH1[4], *pH2[4];
    for (int hd = 0; hd < 4; ++hd) {
        pH1[hd] = alloc_pk(H, H);       repack(hW1[hd], pH1[hd], H, H, 0);
        pH2[hd] = alloc_pk(H, hd2[hd]); repack(hW2[hd], pH2[hd], H, hd2[hd], 0);
    }

    dim3 gGemm(M / 16), bBlk(BLOCK_THREADS);
    dim3 gAgg(Bb * (Nn / 32));
    dim3 gScore(M / 256);

    // ---- GNN layer 1 (din=64) ----
    gemm_bias_act<<<gGemm, bBlk, 0, stream>>>(node, pgW[0], g_b[0], bufA, M, 64, H, 0);
    gnn_scores<<<gScore, bBlk, 0, stream>>>(bufA, g_aW[0], si, sj, M, H);
    gnn_aggregate<<<gAgg, bBlk, 0, stream>>>(bufA, adj, si, sj, g_ab[0], bufB, 1);  // h1 = relu
    // ---- GNN layer 2 ----
    gemm_bias_act<<<gGemm, bBlk, 0, stream>>>(bufB, pgW[1], g_b[1], bufA, M, H, H, 0);
    gnn_scores<<<gScore, bBlk, 0, stream>>>(bufA, g_aW[1], si, sj, M, H);
    gnn_aggregate<<<gAgg, bBlk, 0, stream>>>(bufA, adj, si, sj, g_ab[1], bufC, 1);  // h2 = relu
    // ---- GNN layer 3 (no relu) ----
    gemm_bias_act<<<gGemm, bBlk, 0, stream>>>(bufC, pgW[2], g_b[2], bufA, M, H, H, 0);
    gnn_scores<<<gScore, bBlk, 0, stream>>>(bufA, g_aW[2], si, sj, M, H);
    gnn_aggregate<<<gAgg, bBlk, 0, stream>>>(bufA, adj, si, sj, g_ab[2], bufB, 0);  // h = bufB

    // ---- LSTM layers (sequential, persistent block) ----
    lstm_layer<<<1, bBlk, 0, stream>>>(bufB, pLih[0], pLhh[0], l_bih[0], l_bhh[0], bufC);
    lstm_layer<<<1, bBlk, 0, stream>>>(bufC, pLih[1], pLhh[1], l_bih[1], l_bhh[1], bufA); // lo = bufA

    // ---- MLP heads: op/pa/sk from lo (bufA), no from h (bufB) ----
    float* outp = (float*)d_out;
    size_t off = 0;
    for (int hd = 0; hd < 4; ++hd) {
        const float* src = (hd == 3) ? bufB : bufA;
        gemm_bias_act<<<gGemm, bBlk, 0, stream>>>(src,  pH1[hd], hb1[hd], bufC, M, H, H, 1);
        gemm_bias_act<<<gGemm, bBlk, 0, stream>>>(bufC, pH2[hd], hb2[hd], outp + off, M, H, hd2[hd], 0);
        off += (size_t)M * hd2[hd];
    }
    (void)in_sizes; (void)n_in; (void)out_size; (void)ws_size;
}
